// GE2ELoss_33998961115942
// MI455X (gfx1250) — matrix operand order repaired and checked by reading, not compiled, not run on hardware
//
#include <hip/hip_runtime.h>

#define GS 512            // num_speakers
#define GU 32             // utterances per speaker
#define GD 1024           // embedding dim
#define NROW (GS * GU)    // 16384
#define GEPS 1e-8f

typedef __attribute__((ext_vector_type(16))) _Float16 v16h;
typedef __attribute__((ext_vector_type(4)))  _Float16 v4h;
typedef __attribute__((ext_vector_type(8)))  float    v8f;
typedef __attribute__((ext_vector_type(4)))  float    v4f;

// ---------------------------------------------------------------- zero output
__global__ void k_zero(float* __restrict__ out) { out[0] = 0.0f; }

// ------------------------------------------------- per-speaker centroid sums
__global__ __launch_bounds__(256) void k_centroid_sum(
    const float* __restrict__ emb, float* __restrict__ csum) {
  const int s = blockIdx.x, tid = threadIdx.x;
  const v4f* bp = (const v4f*)(emb + (size_t)s * GU * GD) + tid;
  v4f acc = {0.f, 0.f, 0.f, 0.f};
#pragma unroll 8
  for (int u = 0; u < GU; ++u) acc += bp[u * (GD / 4)];
  ((v4f*)(csum + (size_t)s * GD))[tid] = acc;
}

// -------------------------------------- normalized inclusive centroids (f16)
__global__ __launch_bounds__(256) void k_centroid_hat(
    const float* __restrict__ csum, _Float16* __restrict__ ci_hat) {
  const int s = blockIdx.x, tid = threadIdx.x;
  v4f c = ((const v4f*)(csum + (size_t)s * GD))[tid] * (1.0f / (float)GU);
  float c2 = c.x * c.x + c.y * c.y + c.z * c.z + c.w * c.w;
#pragma unroll
  for (int m = 16; m >= 1; m >>= 1) c2 += __shfl_xor(c2, m, 32);
  __shared__ float sh[8];
  __shared__ float s_inv;
  if ((tid & 31) == 0) sh[tid >> 5] = c2;
  __syncthreads();
  if (tid == 0) {
    float t = 0.f;
#pragma unroll
    for (int i = 0; i < 8; ++i) t += sh[i];
    s_inv = 1.0f / fmaxf(sqrtf(t), GEPS);
  }
  __syncthreads();
  const float inv = s_inv;
  v4h h;
  h.x = (_Float16)(c.x * inv);
  h.y = (_Float16)(c.y * inv);
  h.z = (_Float16)(c.z * inv);
  h.w = (_Float16)(c.w * inv);
  ((v4h*)(ci_hat + (size_t)s * GD))[tid] = h;
}

// --------------- fused: row norms + exact leave-one-out diagonal cosine (f32)
__global__ __launch_bounds__(256) void k_norm_diag(
    const float* __restrict__ emb, const float* __restrict__ csum,
    float* __restrict__ row_norm, float* __restrict__ diag) {
  const int r = blockIdx.x, tid = threadIdx.x, s = r / GU;
  v4f e = ((const v4f*)(emb + (size_t)r * GD))[tid];
  v4f c = ((const v4f*)(csum + (size_t)s * GD))[tid];
  v4f ce = (c - e) * (1.0f / (float)(GU - 1));
  float ss = e.x * e.x + e.y * e.y + e.z * e.z + e.w * e.w;
  float c2 = ce.x * ce.x + ce.y * ce.y + ce.z * ce.z + ce.w * ce.w;
  float nm = ce.x * e.x + ce.y * e.y + ce.z * e.z + ce.w * e.w;
#pragma unroll
  for (int m = 16; m >= 1; m >>= 1) {
    ss += __shfl_xor(ss, m, 32);
    c2 += __shfl_xor(c2, m, 32);
    nm += __shfl_xor(nm, m, 32);
  }
  __shared__ float sh0[8], sh1[8], sh2[8];
  if ((tid & 31) == 0) {
    sh0[tid >> 5] = ss;
    sh1[tid >> 5] = c2;
    sh2[tid >> 5] = nm;
  }
  __syncthreads();
  if (tid == 0) {
    float S0 = 0.f, S1 = 0.f, S2 = 0.f;
#pragma unroll
    for (int i = 0; i < 8; ++i) { S0 += sh0[i]; S1 += sh1[i]; S2 += sh2[i]; }
    const float en = sqrtf(S0);
    row_norm[r] = en;
    diag[r] = S2 / (fmaxf(sqrtf(S1), GEPS) * fmaxf(en, GEPS));
  }
}

// ------------------------------------------------------------ GEMM helpers
// Load one K-step worth of fragments: A raw f32 (16 floats in 4 v4f) and the
// 4 B fragments as single v16h (32B contiguous, 32B aligned -> 2x b128 each).
__device__ __forceinline__ void load_frags(
    const float* __restrict__ ap, const _Float16* __restrict__ bp0, int kb,
    v4f& a0, v4f& a1, v4f& a2, v4f& a3, v16h b[4]) {
  const float* an = ap + kb;
  a0 = *(const v4f*)(an + 0);
  a1 = *(const v4f*)(an + 4);
  a2 = *(const v4f*)(an + 8);
  a3 = *(const v4f*)(an + 12);
#pragma unroll
  for (int t = 0; t < 4; ++t)
    b[t] = *(const v16h*)(bp0 + (size_t)t * 16 * GD + kb);
}

// Convert A to f16 (normalized) and issue the 4 WMMAs for this K-step.
__device__ __forceinline__ void compute_step(
    float inv_n, v4f a0, v4f a1, v4f a2, v4f a3, const v16h b[4], v8f acc[4]) {
  float fa[16];
  *(v4f*)&fa[0] = a0; *(v4f*)&fa[4] = a1;
  *(v4f*)&fa[8] = a2; *(v4f*)&fa[12] = a3;
  v16h afrag;
#pragma unroll
  for (int j = 0; j < 16; ++j) afrag[j] = (_Float16)(fa[j] * inv_n);
#pragma unroll
  for (int t = 0; t < 4; ++t)
    acc[t] = __builtin_amdgcn_wmma_f32_16x16x32_f16(
        false, afrag, false, b[t], (short)0, acc[t], false, false);
}

// ---------------- WMMA GEMM (ê × ĉᵀ) fused with log-softmax cross-entropy
// grid = NROW/16 (16-row strip per block, strip lies within one speaker),
// block = 256 (8 waves); each wave owns 4 N-tiles (64 of the 512 columns).
// K loop unrolled x2 with symmetric ping/pong fragment sets: while one set is
// multiplied, the other set's global loads are in flight (no rotation copies).
__global__ __launch_bounds__(256) void k_gemm_loss(
    const float* __restrict__ emb, const _Float16* __restrict__ ci_hat,
    const float* __restrict__ row_norm, const float* __restrict__ diag,
    const float* __restrict__ wp, const float* __restrict__ bp,
    float* __restrict__ out) {
  __shared__ float sim[16 * GS];   // 32 KB
  __shared__ float rl[16];

  const int tid  = threadIdx.x;
  const int wave = tid >> 5;
  const int lane = tid & 31;
  const int l16  = lane & 15;
  const int hi   = lane >> 4;          // 0 or 1
  const int Mbase = blockIdx.x * 16;
  const int s_own = Mbase / GU;        // whole strip belongs to this speaker
  const int nbase = wave * 64;

  const int arow = Mbase + l16;
  const float inv_n = 1.0f / fmaxf(row_norm[arow], GEPS);
  // A (16x32 f16 layout): per lane, contiguous K = hi*16 .. hi*16+15
  const float* ap = emb + (size_t)arow * GD + hi * 16;
  // B (32x16 f16 layout): lane holds column N, contiguous K = hi*16 .. hi*16+15
  const _Float16* bp0 = ci_hat + (size_t)(nbase + l16) * GD + hi * 16;

  v8f acc[4] = {};

  v4f pa0, pa1, pa2, pa3;   v16h pB[4];   // ping set
  v4f qa0, qa1, qa2, qa3;   v16h qB[4];   // pong set

  load_frags(ap, bp0, 0, pa0, pa1, pa2, pa3, pB);

  for (int kb = 0; kb < GD; kb += 64) {
    load_frags(ap, bp0, kb + 32, qa0, qa1, qa2, qa3, qB);   // kb+32 <= GD-32
    compute_step(inv_n, pa0, pa1, pa2, pa3, pB, acc);
    if (kb + 64 < GD)
      load_frags(ap, bp0, kb + 64, pa0, pa1, pa2, pa3, pB);
    compute_step(inv_n, qa0, qa1, qa2, qa3, qB, acc);
  }

  // D layout: VGPR i, lane L -> row = i + 8*(L/16), col = L%16
#pragma unroll
  for (int t = 0; t < 4; ++t)
#pragma unroll
    for (int i = 0; i < 8; ++i)
      sim[(i + 8 * hi) * GS + nbase + t * 16 + l16] = acc[t][i];
  __syncthreads();

  // ---- per-row logsumexp over 512 logits; 16 threads per row
  const float w = *wp, b = *bp;
  const int row = tid >> 4, sub = tid & 15;
  const float dg = diag[Mbase + row];
  const float ltgt = w * dg + b;       // target (own-speaker) logit

  float lmax = -3.0e38f;
  for (int c = sub; c < GS; c += 16) {
    const float sv = (c == s_own) ? dg : sim[row * GS + c];
    lmax = fmaxf(lmax, w * sv + b);
  }
#pragma unroll
  for (int m = 8; m >= 1; m >>= 1) lmax = fmaxf(lmax, __shfl_xor(lmax, m, 16));

  float lsum = 0.f;
  for (int c = sub; c < GS; c += 16) {
    const float sv = (c == s_own) ? dg : sim[row * GS + c];
    lsum += __expf(w * sv + b - lmax);
  }
#pragma unroll
  for (int m = 8; m >= 1; m >>= 1) lsum += __shfl_xor(lsum, m, 16);

  if (sub == 0) rl[row] = (lmax + __logf(lsum)) - ltgt;  // -logp[target]
  __syncthreads();
  if (tid == 0) {
    float t = 0.f;
#pragma unroll
    for (int i = 0; i < 16; ++i) t += rl[i];
    atomicAdd(out, t * (1.0f / (float)NROW));
  }
}

// ---------------------------------------------------------------------------
extern "C" void kernel_launch(void* const* d_in, const int* in_sizes, int n_in,
                              void* d_out, int out_size, void* d_ws, size_t ws_size,
                              hipStream_t stream) {
  const float* emb = (const float*)d_in[0];
  const float* wp  = (const float*)d_in[1];
  const float* bp  = (const float*)d_in[2];
  float* out = (float*)d_out;

  // workspace layout (~3.2 MB)
  float* csum     = (float*)d_ws;                 // GS*GD f32      (2 MB)
  float* row_norm = csum + (size_t)GS * GD;       // NROW f32       (64 KB)
  float* diag     = row_norm + NROW;              // NROW f32       (64 KB)
  _Float16* ci_hat = (_Float16*)(diag + NROW);    // GS*GD f16      (1 MB)

  k_zero<<<1, 1, 0, stream>>>(out);
  k_centroid_sum<<<GS, 256, 0, stream>>>(emb, csum);
  k_centroid_hat<<<GS, 256, 0, stream>>>(csum, ci_hat);
  k_norm_diag<<<NROW, 256, 0, stream>>>(emb, csum, row_norm, diag);
  k_gemm_loss<<<NROW / 16, 256, 0, stream>>>(emb, ci_hat, row_norm, diag,
                                             wp, bp, out);
}
